// QuantizedLinear_34016140984680
// MI455X (gfx1250) — compile-verified, matrix operand order checked by reading, not run on hardware
//
#include <hip/hip_runtime.h>

// Problem constants (match reference)
#define TOKENS 8192
#define INF    4096
#define OUTF   11008
#define NG     32        // INF / 128 groups

#define BM 128
#define BN 256
#define BK 32
#define LDA 40           // padded LDS row stride in halves (80 B)
#define NIT (INF / BK)   // 128 k-steps

typedef __attribute__((ext_vector_type(16))) _Float16 v16h;
typedef __attribute__((ext_vector_type(8)))  _Float16 v8h;
typedef __attribute__((ext_vector_type(8)))  float    v8f;

__global__ __launch_bounds__(256)
void gptq_wmma_gemm(const float* __restrict__ x,
                    const int*   __restrict__ qw,
                    const float* __restrict__ scales,
                    const float* __restrict__ zeros,
                    const float* __restrict__ bias,
                    float* __restrict__ out)
{
    __shared__ _Float16 As[2][BM][LDA];   // x tile (f16)           : 20 KB
    __shared__ _Float16 Bs[2][BN][LDA];   // dequant W tile (f16)   : 40 KB

    const int tid  = threadIdx.x;
    const int lane = tid & 31;
    const int wave = tid >> 5;
    const int wm   = wave & 1;     // 2 wave rows  -> 64 output rows each
    const int wn   = wave >> 1;    // 4 wave cols  -> 64 output cols each

    const int rowBase = blockIdx.y * BM;   // token dim
    const int colBase = blockIdx.x * BN;   // out-feature dim

    // ---- global-load thread mappings ----
    // x: 2 threads per tile row (128 rows), 16 floats each
    const int xRow = tid >> 1;
    const int xCol = (tid & 1) * 16;
    // qweight: 1 thread per tile row (256 rows), 16 int32 each
    const int qRow = tid;

    const float* xg = x      + (size_t)(rowBase + xRow) * INF;
    const int*   qg = qw     + (size_t)(colBase + qRow) * (INF / 2);
    const float* sg = scales + (size_t)(colBase + qRow) * NG;
    const float* zg = zeros  + (size_t)(colBase + qRow) * NG;

    float4 xr[4];
    int4   qr[4];
    float  sc, zp;

    // global -> registers for k-step starting at k0
    auto gload = [&](int k0) {
        const float4* xp = (const float4*)(xg + k0 + xCol);
        xr[0] = xp[0]; xr[1] = xp[1]; xr[2] = xp[2]; xr[3] = xp[3];
        const int4* qp = (const int4*)(qg + (k0 >> 1));
        qr[0] = qp[0]; qr[1] = qp[1]; qr[2] = qp[2]; qr[3] = qp[3];
        const int g = k0 >> 7;             // whole k-step lies in one group
        sc = sg[g];
        zp = zg[g];
    };

    // registers -> LDS (convert fp32->f16, dequant int4->f16)
    auto lstore = [&](int buf) {
        const float* xf = (const float*)&xr[0];
        v8h h0, h1;
#pragma unroll
        for (int i = 0; i < 8; ++i) {
            h0[i] = (_Float16)xf[i];
            h1[i] = (_Float16)xf[8 + i];
        }
        *(v8h*)&As[buf][xRow][xCol]     = h0;
        *(v8h*)&As[buf][xRow][xCol + 8] = h1;

        const int* qi = (const int*)&qr[0];
#pragma unroll
        for (int c = 0; c < 4; ++c) {      // 4 chunks of 4 int32 -> 8 halves
            v8h w;
#pragma unroll
            for (int j = 0; j < 4; ++j) {
                const int q  = qi[4 * c + j];
                const float lo = (float)(q & 15);        // col 2i   = low nibble
                const float hi = (float)((q >> 4) & 15); // col 2i+1 = high nibble
                w[2 * j]     = (_Float16)((lo - zp) * sc);
                w[2 * j + 1] = (_Float16)((hi - zp) * sc);
            }
            *(v8h*)&Bs[buf][qRow][8 * c] = w;
        }
    };

    v8f acc[4][4] = {};

    // one k-step of WMMAs from LDS buffer `buf`
    auto compute = [&](int buf) {
        const int r  = lane & 15;
        const int kb = (lane >> 4) * 8;    // ISA 16-bit A/B lane->K pattern
        v16h af[4];
#pragma unroll
        for (int mt = 0; mt < 4; ++mt) {
            const _Float16* p = &As[buf][wm * 64 + mt * 16 + r][kb];
            v8h lo = *(const v8h*)p;
            v8h hi = *(const v8h*)(p + 16);
            af[mt] = __builtin_shufflevector(lo, hi, 0,1,2,3,4,5,6,7,8,9,10,11,12,13,14,15);
        }
#pragma unroll
        for (int nt = 0; nt < 4; ++nt) {   // stream B fragments: short live range
            const _Float16* p = &Bs[buf][wn * 64 + nt * 16 + r][kb];
            v8h lo = *(const v8h*)p;
            v8h hi = *(const v8h*)(p + 16);
            v16h bf = __builtin_shufflevector(lo, hi, 0,1,2,3,4,5,6,7,8,9,10,11,12,13,14,15);
#pragma unroll
            for (int mt = 0; mt < 4; ++mt)
                acc[mt][nt] = __builtin_amdgcn_wmma_f32_16x16x32_f16(
                    false, af[mt], false, bf, (short)0, acc[mt][nt], false, false);
        }
    };

    // ---- software-pipelined main loop (double-buffered LDS) ----
    gload(0);
    lstore(0);
    __syncthreads();
    for (int it = 0; it < NIT; ++it) {
        if (it + 1 < NIT) gload((it + 1) * BK);
        compute(it & 1);
        if (it + 1 < NIT) {
            lstore((it + 1) & 1);
            __syncthreads();
        }
    }

    // ---- epilogue: add bias, store (C/D layout: m=(lane>=16?8:0)+j, n=lane&15) ----
    const int n  = lane & 15;
    const int mh = (lane >> 4) << 3;
#pragma unroll
    for (int nt = 0; nt < 4; ++nt) {
        const int o  = colBase + wn * 64 + nt * 16 + n;
        const float bv = bias[o];
#pragma unroll
        for (int mt = 0; mt < 4; ++mt) {
            const int t0 = rowBase + wm * 64 + mt * 16 + mh;
            float* op = out + (size_t)t0 * OUTF + o;
#pragma unroll
            for (int j = 0; j < 8; ++j)
                op[(size_t)j * OUTF] = acc[mt][nt][j] + bv;
        }
    }
}

extern "C" void kernel_launch(void* const* d_in, const int* in_sizes, int n_in,
                              void* d_out, int out_size, void* d_ws, size_t ws_size,
                              hipStream_t stream) {
    (void)in_sizes; (void)n_in; (void)out_size; (void)d_ws; (void)ws_size;
    const float* x      = (const float*)d_in[0];
    const int*   qw     = (const int*)  d_in[1];
    const float* scales = (const float*)d_in[2];
    const float* zeros  = (const float*)d_in[3];
    const float* bias   = (const float*)d_in[4];
    float* out = (float*)d_out;

    dim3 grid(OUTF / BN, TOKENS / BM);   // 43 x 64
    dim3 block(256);
    gptq_wmma_gemm<<<grid, block, 0, stream>>>(x, qw, scales, zeros, bias, out);
}
